// ATACSplitPool_46961172415291
// MI455X (gfx1250) — matrix operand order repaired and checked by reading, not compiled, not run on hardware
//
#include <hip/hip_runtime.h>
#include <hip/hip_bf16.h>

typedef __attribute__((ext_vector_type(16))) _Float16 v16h;
typedef __attribute__((ext_vector_type(8)))  float    v8f;

#define PATCH 25
#define BN_EPS 1e-5f

// ---------------------------------------------------------------------------
// 0. zero fill (dword granularity)
__global__ void zero_kernel(float* p, long long n) {
    long long i = (long long)blockIdx.x * blockDim.x + threadIdx.x;
    if (i < n) p[i] = 0.0f;
}

// ---------------------------------------------------------------------------
// 1. prefix sums over ragged segment table (tiny, serial)
__global__ void prefix_kernel(const int* __restrict__ peak_split,
                              const int* __restrict__ n_peaks,
                              int n_seg, int B,
                              int* __restrict__ seg_start,
                              int* __restrict__ samp_off) {
    if (threadIdx.x == 0 && blockIdx.x == 0) {
        int acc = 0;
        for (int i = 0; i < n_seg; ++i) { seg_start[i] = acc; acc += peak_split[i]; }
        int s = 0;
        for (int b = 0; b < B; ++b) { samp_off[b] = s; s += n_peaks[b] + 1; }
        samp_off[B] = s;
    }
}

// ---------------------------------------------------------------------------
// 2. pad joint_w (Cout, Cin, 3) -> wpad[co_pad][k*Cin + ci]  (f16, zero-padded)
__global__ void wpad_kernel(const float* __restrict__ jw, _Float16* __restrict__ wpad,
                            int Cout, int CoutPad, int Cin) {
    int idx = blockIdx.x * blockDim.x + threadIdx.x;
    int Kdim = 3 * Cin;
    if (idx >= CoutPad * Kdim) return;
    int co = idx / Kdim;
    int K  = idx - co * Kdim;
    int k  = K / Cin;
    int ci = K - k * Cin;
    float v = (co < Cout) ? jw[(co * Cin + ci) * 3 + k] : 0.0f;
    wpad[idx] = (_Float16)v;
}

// ---------------------------------------------------------------------------
// 3. atac: log10(x+1) then maxpool(25)  ->  atac_p (B*Lp)
__global__ void atac_pool_kernel(const float* __restrict__ atac, float* __restrict__ atac_p,
                                 int B, int L, int Lp) {
    int i = blockIdx.x * blockDim.x + threadIdx.x;
    if (i >= B * Lp) return;
    int b = i / Lp, lp = i - b * Lp;
    const float* src = atac + (size_t)b * L + lp * PATCH;
    float mx = src[0];
    #pragma unroll
    for (int p = 1; p < PATCH; ++p) mx = fmaxf(mx, src[p]);
    atac_p[i] = log10f(mx + 1.0f);  // log10 monotone -> pool-then-log == log-then-pool
}

// ---------------------------------------------------------------------------
// 4. atac conv (1 -> Cout, K=3, same) -> a_raw (B, Cout, Lp)
__global__ void atac_conv_kernel(const float* __restrict__ atac_p, const float* __restrict__ aw,
                                 float* __restrict__ a_raw, int B, int Cout, int Lp) {
    int i = blockIdx.x * blockDim.x + threadIdx.x;
    if (i >= B * Cout * Lp) return;
    int lp = i % Lp;
    int t  = i / Lp;
    int co = t % Cout;
    int b  = t / Cout;
    float s = 0.0f;
    #pragma unroll
    for (int k = 0; k < 3; ++k) {
        int l = lp + k - 1;
        float v = (l >= 0 && l < Lp) ? atac_p[b * Lp + l] : 0.0f;
        s += v * aw[co * 3 + k];
    }
    a_raw[i] = s;
}

// ---------------------------------------------------------------------------
// 5. BN stats: per-channel mean / invstd over (B, Lp). src layout (B, C, Lp).
__global__ void bn_stats_kernel(const float* __restrict__ src, float* __restrict__ mean,
                                float* __restrict__ invstd, int B, int C, int Lp) {
    __shared__ float ssum[256];
    __shared__ float ssq[256];
    int c = blockIdx.x;
    int tid = threadIdx.x;
    int N = B * Lp;
    float s = 0.0f, q = 0.0f;
    for (int i = tid; i < N; i += blockDim.x) {
        int b = i / Lp, lp = i - b * Lp;
        float v = src[((size_t)b * C + c) * Lp + lp];
        s += v; q += v * v;
    }
    ssum[tid] = s; ssq[tid] = q;
    __syncthreads();
    for (int off = 128; off > 0; off >>= 1) {
        if (tid < off) { ssum[tid] += ssum[tid + off]; ssq[tid] += ssq[tid + off]; }
        __syncthreads();
    }
    if (tid == 0) {
        float m = ssum[0] / (float)N;
        float var = ssq[0] / (float)N - m * m;
        mean[c] = m;
        invstd[c] = rsqrtf(var + BN_EPS);
    }
}

// ---------------------------------------------------------------------------
// 6. BN apply + relu for atac branch, write f16 into hT channels [D, D+Cout)
//    hT rows are offset by +1 (front halo row).
__global__ void bn_apply_atac_kernel(const float* __restrict__ a_raw,
                                     const float* __restrict__ mean, const float* __restrict__ invstd,
                                     _Float16* __restrict__ hT,
                                     int B, int Cout, int Lp, int LpP, int D, int Cin) {
    int i = blockIdx.x * blockDim.x + threadIdx.x;
    if (i >= B * Cout * Lp) return;
    int lp = i % Lp;
    int t  = i / Lp;
    int c  = t % Cout;
    int b  = t / Cout;
    float v = (a_raw[i] - mean[c]) * invstd[c];
    v = fmaxf(v, 0.0f);
    hT[((size_t)(b * LpP + lp + 1)) * Cin + D + c] = (_Float16)v;
}

// ---------------------------------------------------------------------------
// 7. x maxpool(25): x (B,L,D) -> hT[b][lp+1][0..D) f16   (the big 511MB read #1)
__global__ void xpool_kernel(const float* __restrict__ x, _Float16* __restrict__ hT,
                             int B, int L, int Lp, int LpP, int D, int Cin) {
    int blk = blockIdx.x;                 // b*Lp + lp
    int b = blk / Lp, lp = blk - b * Lp;
    const float* base = x + ((size_t)b * L + lp * PATCH) * D;
    _Float16* dst = hT + ((size_t)(b * LpP + lp + 1)) * Cin;
    for (int d = threadIdx.x; d < D; d += blockDim.x) {
        float mx = base[d];
        #pragma unroll
        for (int p = 1; p < PATCH; ++p) mx = fmaxf(mx, base[(size_t)p * D + d]);
        dst[d] = (_Float16)mx;
    }
}

// ---------------------------------------------------------------------------
// 8. joint conv as implicit GEMM with WMMA f16 (branchless: hT has zero halo
//    rows, so every B-operand load is in-bounds).
//    h2raw[b][co][lp] = sum_{k,ci} hT[b][lp+k-1][ci] * wpad[co][k*Cin+ci]
__global__ void __launch_bounds__(32)
wmma_conv_kernel(const _Float16* __restrict__ hT, const _Float16* __restrict__ wpad,
                 float* __restrict__ h2raw, int B, int Cout, int Lp, int LpP, int Cin) {
    const int lane = threadIdx.x;
    const int g = lane >> 4;     // half-wave group
    const int m = lane & 15;     // A row (out-channel) / B column (lp)
    const int lp0 = blockIdx.x * 16;
    const int co0 = blockIdx.y * 16;
    const int b   = blockIdx.z;
    const int Kdim = 3 * Cin;
    const int cTiles = Cin / 32;         // Cin is a multiple of 32 (800)

    v8f acc = {};
    const _Float16* abase = wpad + (size_t)(co0 + m) * Kdim + g * 16;

    #pragma unroll
    for (int k = 0; k < 3; ++k) {
        // B row for this lane is fixed per k-phase; halo makes it always valid:
        // logical lp index = lp0 + m + k - 1, stored at row (lp0 + m + k).
        const _Float16* brow = hT + ((size_t)(b * LpP + lp0 + m + k)) * Cin + g * 16;
        const _Float16* arow = abase + k * Cin;

        for (int ct = 0; ct < cTiles; ++ct) {
            // A: weights tile, lane holds K offsets {g*16 .. g*16+15}
            union { v16h v; uint4 q[2]; } A;
            A.q[0] = *(const uint4*)(arow);
            A.q[1] = *(const uint4*)(arow + 8);
            arow += 32;

            // B: activation tile, lane holds ci offsets {g*16 .. g*16+15}
            union { v16h v; uint4 q[2]; } Bv;
            Bv.q[0] = *(const uint4*)(brow);
            Bv.q[1] = *(const uint4*)(brow + 8);
            __builtin_prefetch(brow + 32, 0, 1);   // -> global_prefetch_b8
            brow += 32;

            acc = __builtin_amdgcn_wmma_f32_16x16x32_f16(
                /*neg_a=*/false, A.v, /*neg_b=*/false, Bv.v,
                /*c_mod=*/(short)0, acc, /*reuse_a=*/false, /*reuse_b=*/false);
        }
    }

    // D layout: lane l, reg r -> co = co0 + r + 8*g, lp = lp0 + (l%16)
    union { v8f v; float f[8]; } R;
    R.v = acc;
    const int lp = lp0 + m;
    if (lp < Lp) {
        #pragma unroll
        for (int r = 0; r < 8; ++r) {
            int co = co0 + r + 8 * g;
            if (co < Cout) h2raw[((size_t)b * Cout + co) * Lp + lp] = R.f[r];
        }
    }
}

// ---------------------------------------------------------------------------
// 9. BN apply + relu for joint branch (fp32 h2)
__global__ void bn_apply_joint_kernel(const float* __restrict__ h2raw,
                                      const float* __restrict__ mean, const float* __restrict__ invstd,
                                      float* __restrict__ h2, int B, int Cout, int Lp) {
    int i = blockIdx.x * blockDim.x + threadIdx.x;
    if (i >= B * Cout * Lp) return;
    int c = (i / Lp) % Cout;
    h2[i] = fmaxf((h2raw[i] - mean[c]) * invstd[c], 0.0f);
}

// ---------------------------------------------------------------------------
// 10. x_region: ragged mean of raw x rows per kept chunk (big 511MB read #2)
__global__ void xregion_kernel(const float* __restrict__ x,
                               const int* __restrict__ peak_split,
                               const int* __restrict__ n_peaks,
                               const int* __restrict__ seg_start,
                               const int* __restrict__ samp_off,
                               float* __restrict__ out,
                               int maxP, int D, int Cin) {
    int bj = blockIdx.x;
    int b = bj / maxP, j = bj - b * maxP;
    int d = blockIdx.y * blockDim.x + threadIdx.x;
    if (d >= D) return;
    if (j >= n_peaks[b]) return;                    // output already zeroed
    int seg = samp_off[b] + j;
    int start = seg_start[seg];                     // global flat row into (B*L, D)
    int len = peak_split[seg];
    float s = 0.0f;
    const float* p = x + (size_t)start * D + d;
    for (int r = 0; r < len; ++r) s += p[(size_t)r * D];
    float cnt = (float)max(len, 1);
    out[((size_t)b * maxP + j) * Cin + d] = s / cnt;
}

// ---------------------------------------------------------------------------
// 11. joint ragged mean over patches + log2(+1)
__global__ void joint_segmean_kernel(const float* __restrict__ h2,
                                     const int* __restrict__ peak_split,
                                     const int* __restrict__ n_peaks,
                                     const int* __restrict__ seg_start,
                                     const int* __restrict__ samp_off,
                                     float* __restrict__ out,
                                     int maxP, int Cout, int Lp, int D, int Cin) {
    int bj = blockIdx.x;
    int b = bj / maxP, j = bj - b * maxP;
    int c = threadIdx.x;
    if (c >= Cout) return;
    if (j >= n_peaks[b]) return;                    // output already zeroed
    int seg = samp_off[b] + j;
    int pstart = seg_start[seg] / PATCH;            // flat patch index into (B*Lp)
    int plen = peak_split[seg] / PATCH;
    int bb = pstart / Lp, lp0 = pstart - bb * Lp;
    const float* p = h2 + ((size_t)bb * Cout + c) * Lp + lp0;
    float s = 0.0f;
    for (int i = 0; i < plen; ++i) s += p[i];
    float mval = s / (float)max(plen, 1);
    out[((size_t)b * maxP + j) * Cin + D + c] = log2f(mval + 1.0f);
}

// ---------------------------------------------------------------------------
extern "C" void kernel_launch(void* const* d_in, const int* in_sizes, int n_in,
                              void* d_out, int out_size, void* d_ws, size_t ws_size,
                              hipStream_t stream) {
    const float* x       = (const float*)d_in[0];
    const float* atac    = (const float*)d_in[1];
    const float* atac_w  = (const float*)d_in[2];
    const float* joint_w = (const float*)d_in[3];
    const int*   psplit  = (const int*)d_in[4];
    const int*   npeaks  = (const int*)d_in[5];
    float* out = (float*)d_out;

    // derive shapes from sizes (no device reads)
    const int B    = in_sizes[5];                 // 8
    const int Cout = in_sizes[2] / 3;             // 161
    const int Cin  = in_sizes[3] / (Cout * 3);    // 800
    const int D    = Cin - Cout;                  // 639
    const int L    = in_sizes[1] / B;             // 25000
    const int Lp   = L / PATCH;                   // 1000
    const int LpP  = Lp + 32;                     // halo-padded rows per batch
    const int nseg = in_sizes[4];
    const int maxP = out_size / (B * Cin);        // 200
    const int CoutPad = ((Cout + 15) / 16) * 16;  // 176
    const int Kdim = 3 * Cin;

    // workspace partition (256B aligned)
    char* w = (char*)d_ws;
    size_t off = 0;
    auto alloc = [&](size_t bytes) { void* p = w + off; off = (off + bytes + 255) & ~(size_t)255; return p; };
    _Float16* hT     = (_Float16*)alloc((size_t)B * LpP * Cin * sizeof(_Float16));
    _Float16* wpad   = (_Float16*)alloc((size_t)CoutPad * Kdim * sizeof(_Float16));
    float* atac_p    = (float*)alloc((size_t)B * Lp * sizeof(float));
    float* a_raw     = (float*)alloc((size_t)B * Cout * Lp * sizeof(float));
    float* h2raw     = (float*)alloc((size_t)B * Cout * Lp * sizeof(float));
    float* h2        = (float*)alloc((size_t)B * Cout * Lp * sizeof(float));
    float* meanA     = (float*)alloc(Cout * sizeof(float));
    float* invA      = (float*)alloc(Cout * sizeof(float));
    float* meanJ     = (float*)alloc(Cout * sizeof(float));
    float* invJ      = (float*)alloc(Cout * sizeof(float));
    int* seg_start   = (int*)alloc((size_t)nseg * sizeof(int));
    int* samp_off    = (int*)alloc((size_t)(B + 1) * sizeof(int));
    (void)ws_size;

    const int T = 256;
    // 0. zero output (masked peaks stay zero) and hT (halo rows must be zero)
    zero_kernel<<<(out_size + T - 1) / T, T, 0, stream>>>(out, out_size);
    long long hTdw = (long long)B * LpP * Cin / 2;   // f16 buffer as dwords
    zero_kernel<<<(unsigned)((hTdw + T - 1) / T), T, 0, stream>>>((float*)hT, hTdw);
    // 1. ragged prefix tables
    prefix_kernel<<<1, 32, 0, stream>>>(psplit, npeaks, nseg, B, seg_start, samp_off);
    // 2. weight pad/convert
    wpad_kernel<<<((CoutPad * Kdim) + T - 1) / T, T, 0, stream>>>(joint_w, wpad, Cout, CoutPad, Cin);
    // 3. atac pool + log10
    atac_pool_kernel<<<(B * Lp + T - 1) / T, T, 0, stream>>>(atac, atac_p, B, L, Lp);
    // 4. atac conv
    atac_conv_kernel<<<(B * Cout * Lp + T - 1) / T, T, 0, stream>>>(atac_p, atac_w, a_raw, B, Cout, Lp);
    // 5. atac BN stats + apply -> hT upper channels
    bn_stats_kernel<<<Cout, T, 0, stream>>>(a_raw, meanA, invA, B, Cout, Lp);
    bn_apply_atac_kernel<<<(B * Cout * Lp + T - 1) / T, T, 0, stream>>>(a_raw, meanA, invA, hT, B, Cout, Lp, LpP, D, Cin);
    // 6. x maxpool -> hT lower channels
    xpool_kernel<<<B * Lp, T, 0, stream>>>(x, hT, B, L, Lp, LpP, D, Cin);
    // 7. joint conv via WMMA (branchless inner loop)
    dim3 gg((Lp + 15) / 16, CoutPad / 16, B);
    wmma_conv_kernel<<<gg, 32, 0, stream>>>(hT, wpad, h2raw, B, Cout, Lp, LpP, Cin);
    // 8. joint BN stats + apply
    bn_stats_kernel<<<Cout, T, 0, stream>>>(h2raw, meanJ, invJ, B, Cout, Lp);
    bn_apply_joint_kernel<<<(B * Cout * Lp + T - 1) / T, T, 0, stream>>>(h2raw, meanJ, invJ, h2, B, Cout, Lp);
    // 9. ragged outputs
    dim3 gx(B * maxP, (D + T - 1) / T);
    xregion_kernel<<<gx, T, 0, stream>>>(x, psplit, npeaks, seg_start, samp_off, out, maxP, D, Cin);
    joint_segmean_kernel<<<B * maxP, 192, 0, stream>>>(h2, psplit, npeaks, seg_start, samp_off, out, maxP, Cout, Lp, D, Cin);
}
